// HungarianMatcher_34119220200088
// MI455X (gfx1250) — matrix operand order repaired and checked by reading, not compiled, not run on hardware
//
#include <hip/hip_runtime.h>
#include <hip/hip_bf16.h>

typedef __attribute__((ext_vector_type(16))) _Float16 v16h;
typedef __attribute__((ext_vector_type(8)))  _Float16 v8h;
typedef __attribute__((ext_vector_type(4)))  _Float16 v4h;
typedef __attribute__((ext_vector_type(8)))  float    v8f;

#define NQTOT 1200      // bs*nq = 4*300
#define NC    92
#define TGT   100
#define TPAD  112       // 7 * 16 (N padded to WMMA tiles)
#define HW    16384     // 128*128
#define H0    512
#define NCHUNK 256      // K chunks of 64 (16384/64)

// fast sigmoid: 1/(1+exp(-x)) with native exp2 + rcp (we truncate to f16 anyway)
__device__ __forceinline__ float fsig(float x) {
    return __builtin_amdgcn_rcpf(1.0f + __builtin_amdgcn_exp2f(x * -1.44269504088896f));
}

// ---------------------------------------------------------------------------
// Kernel 1: downsample tgt_mask [100,512,512] -> tm f16 [112,16384] (+zeros pad)
// half-pixel bilinear, scale 4 => average of the 4 center pixels.
// Also per-row sums tmsum[112].
// ---------------------------------------------------------------------------
__global__ void hm_downsample_kernel(const float* __restrict__ tgt_mask,
                                     _Float16* __restrict__ tm16,
                                     float* __restrict__ tmsum) {
    __shared__ float sred[256];
    const int t   = blockIdx.x;
    const int tid = threadIdx.x;
    float lsum = 0.0f;
    if (t < TGT) {
        const float* S = tgt_mask + (size_t)t * H0 * H0;
        for (int p = tid; p < HW; p += 256) {
            const int y = p >> 7, x = p & 127;
            const int iy = 4 * y + 1, ix = 4 * x + 1;
            const float v = 0.25f * (S[iy * H0 + ix]       + S[iy * H0 + ix + 1] +
                                     S[(iy + 1) * H0 + ix] + S[(iy + 1) * H0 + ix + 1]);
            tm16[(size_t)t * HW + p] = (_Float16)v;
            lsum += v;
        }
    } else {
        for (int p = tid; p < HW; p += 256)
            tm16[(size_t)t * HW + p] = (_Float16)0.0f;
    }
    sred[tid] = lsum;
    __syncthreads();
    for (int s = 128; s > 0; s >>= 1) {
        if (tid < s) sred[tid] += sred[tid + s];
        __syncthreads();
    }
    if (tid == 0) tmsum[t] = sred[0];
}

// ---------------------------------------------------------------------------
// Kernel 2: scalar costs (class + bbox L1 + giou) -> d_out[i*100+j]
// One block per query row i (1200 blocks). Softmax denom via LDS reduction.
// ---------------------------------------------------------------------------
__global__ void hm_scalar_cost_kernel(const float* __restrict__ logits,
                                      const float* __restrict__ boxes,
                                      const int*   __restrict__ tgt_ids,
                                      const float* __restrict__ tgt_bbox,
                                      float* __restrict__ out) {
    __shared__ float sred[128];
    const int i   = blockIdx.x;
    const int tid = threadIdx.x;
    const float* lrow = logits + (size_t)i * NC;

    // row max
    float lv = (tid < NC) ? lrow[tid] : -3.4e38f;
    sred[tid] = lv;
    __syncthreads();
    for (int s = 64; s > 0; s >>= 1) {
        if (tid < s) sred[tid] = fmaxf(sred[tid], sred[tid + s]);
        __syncthreads();
    }
    const float mx = sred[0];
    __syncthreads();
    // sum exp
    sred[tid] = (tid < NC) ? __expf(lv - mx) : 0.0f;
    __syncthreads();
    for (int s = 64; s > 0; s >>= 1) {
        if (tid < s) sred[tid] += sred[tid + s];
        __syncthreads();
    }
    const float Z = sred[0];

    if (tid < TGT) {
        const int j = tid;
        const int cid = tgt_ids[j];
        const float cls = -__expf(lrow[cid] - mx) / Z;
        const float* ob = boxes + (size_t)i * 4;
        const float* tb = tgt_bbox + (size_t)j * 4;
        const float ocx = ob[0], ocy = ob[1], ow = ob[2], oh = ob[3];
        const float tcx = tb[0], tcy = tb[1], tw = tb[2], th = tb[3];
        const float l1 = fabsf(ocx - tcx) + fabsf(ocy - tcy) +
                         fabsf(ow - tw) + fabsf(oh - th);
        const float x1a = ocx - 0.5f * ow, y1a = ocy - 0.5f * oh;
        const float x2a = ocx + 0.5f * ow, y2a = ocy + 0.5f * oh;
        const float x1b = tcx - 0.5f * tw, y1b = tcy - 0.5f * th;
        const float x2b = tcx + 0.5f * tw, y2b = tcy + 0.5f * th;
        const float a1 = (x2a - x1a) * (y2a - y1a);
        const float a2 = (x2b - x1b) * (y2b - y1b);
        const float iw = fmaxf(0.0f, fminf(x2a, x2b) - fmaxf(x1a, x1b));
        const float ih = fmaxf(0.0f, fminf(y2a, y2b) - fmaxf(y1a, y1b));
        const float inter = iw * ih;
        const float uni = a1 + a2 - inter;
        const float iou = inter / uni;
        const float ew = fmaxf(x2a, x2b) - fminf(x1a, x1b);
        const float eh = fmaxf(y2a, y2b) - fminf(y1a, y1b);
        const float enc = fmaxf(0.0f, ew) * fmaxf(0.0f, eh);
        const float giou = iou - (enc - uni) / enc;
        out[(size_t)i * TGT + j] = l1 + cls - giou;
    }
}

// ---------------------------------------------------------------------------
// Kernel 3: producer/consumer WMMA dice-mask GEMM.
// One 256-thread block (8 waves) per 16-row M-tile, covering all 7 N-tiles.
// All 8 waves cooperatively sigmoid+convert one 16x64 A chunk into
// double-buffered LDS (sigmoid computed ONCE per element; pred_masks crosses
// L2 once). Waves 0..6 each own a 16x16 output tile and consume A from LDS
// (ds_load_b128) + B from the L2-resident tm buffer.
// ---------------------------------------------------------------------------
#define LROW 72  // padded LDS row (f16) to break the 128B bank stride

__device__ __forceinline__ v16h hm_lds_a_frag(const _Float16* p) {
    v8h lo = *reinterpret_cast<const v8h*>(p);
    v8h hi = *reinterpret_cast<const v8h*>(p + 8);
    v16h a;
#pragma unroll
    for (int e = 0; e < 8; ++e) { a[e] = lo[e]; a[e + 8] = hi[e]; }
    return a;
}

__global__ void __launch_bounds__(256)
hm_mask_gemm_kernel(const float* __restrict__ pred_masks,
                    const _Float16* __restrict__ tm16,
                    const float* __restrict__ tmsum,
                    float* __restrict__ out) {
    __shared__ __align__(16) _Float16 lA[2][16][LROW];
    __shared__ float smrow[16];

    const int tid   = threadIdx.x;
    const int mTile = blockIdx.x;          // 0..74
    const int lane  = tid & 31;
    const int w     = tid >> 5;            // wave 0..7 (0..6 compute)
    const int laneHigh = lane >> 4;
    const int m     = lane & 15;

    // producer mapping: 16 rows x 64 K per chunk, 4 f32 per thread
    const int prow = tid >> 4;             // 0..15
    const int pcol = (tid & 15) << 2;      // 0,4,..,60
    const float* Ap = pred_masks + (size_t)(mTile * 16 + prow) * HW + pcol;

    // consumer WMMA fragment offsets (16-bit A/B VGPR layouts)
    const int kaOff = laneHigh ? 16 : 0;   // A: lanes 0-15 K{0..15}; 16-31 K{16..31}
    const int kbOff = laneHigh ? 16 : 0;   // B: lane=N, contiguous K run per half-wave
    const int wb = (w < 7) ? w : 6;        // keep address computation in-bounds
    const _Float16* Brow = tm16 + (size_t)(wb * 16 + m) * HW;

    float rs = 0.0f;                       // sigmoid row-sum partial (row prow)
    v8f c0 = {}, c1 = {};

    // ---- produce chunk 0 into buffer 0 ----
    {
        const float4 x = *reinterpret_cast<const float4*>(Ap);
        const float s0 = fsig(x.x), s1 = fsig(x.y), s2 = fsig(x.z), s3 = fsig(x.w);
        rs += (s0 + s1) + (s2 + s3);
        v4h h; h[0] = (_Float16)s0; h[1] = (_Float16)s1;
               h[2] = (_Float16)s2; h[3] = (_Float16)s3;
        *reinterpret_cast<v4h*>(&lA[0][prow][pcol]) = h;
    }
    __syncthreads();

    for (int it = 0; it < NCHUNK; ++it) {
        const int buf   = it & 1;
        const int kBase = it * 64;

        // ---- consume chunk it (waves 0..6; wave-uniform branch, EXEC all-1s) ----
        if (w < 7) {
            const v16h a0 = hm_lds_a_frag(&lA[buf][m][kaOff]);
            const v16h b0 = *reinterpret_cast<const v16h*>(Brow + kBase + kbOff);
            c0 = __builtin_amdgcn_wmma_f32_16x16x32_f16(false, a0, false, b0,
                                                        (short)0, c0, false, false);
            const v16h a1 = hm_lds_a_frag(&lA[buf][m][32 + kaOff]);
            const v16h b1 = *reinterpret_cast<const v16h*>(Brow + kBase + 32 + kbOff);
            c1 = __builtin_amdgcn_wmma_f32_16x16x32_f16(false, a1, false, b1,
                                                        (short)0, c1, false, false);
        }

        // ---- produce chunk it+1 into the other buffer ----
        if (it + 1 < NCHUNK) {
            __builtin_prefetch(Ap + (size_t)(it + 3) * 64, 0, 1);
            const float4 x = *reinterpret_cast<const float4*>(Ap + (size_t)(it + 1) * 64);
            const float s0 = fsig(x.x), s1 = fsig(x.y), s2 = fsig(x.z), s3 = fsig(x.w);
            rs += (s0 + s1) + (s2 + s3);
            v4h h; h[0] = (_Float16)s0; h[1] = (_Float16)s1;
                   h[2] = (_Float16)s2; h[3] = (_Float16)s3;
            *reinterpret_cast<v4h*>(&lA[buf ^ 1][prow][pcol]) = h;
        }
        __syncthreads();
    }

    // ---- sigmoid row sums: reduce 16 threads per row (within 16-lane groups) ----
    rs += __shfl_xor(rs, 8, 32);
    rs += __shfl_xor(rs, 4, 32);
    rs += __shfl_xor(rs, 2, 32);
    rs += __shfl_xor(rs, 1, 32);
    if ((tid & 15) == 0) smrow[prow] = rs;
    __syncthreads();

    // ---- finalize: dice cost added onto scalar costs from kernel 2 ----
    if (w < 7) {
        const v8f c = c0 + c1;
        const int n = w * 16 + m;
        if (n < TGT) {
            const float tsum = tmsum[n];
#pragma unroll
            for (int r = 0; r < 8; ++r) {
                const int ml = r + (laneHigh ? 8 : 0);
                const float dice = 1.0f - (2.0f * c[r] + 1.0f) / (smrow[ml] + tsum + 1.0f);
                const size_t o = (size_t)(mTile * 16 + ml) * TGT + n;
                out[o] = out[o] + dice;
            }
        }
    }
}

// ---------------------------------------------------------------------------
extern "C" void kernel_launch(void* const* d_in, const int* in_sizes, int n_in,
                              void* d_out, int out_size, void* d_ws, size_t ws_size,
                              hipStream_t stream) {
    const float* pred_logits = (const float*)d_in[0];  // [4,300,92]
    const float* pred_boxes  = (const float*)d_in[1];  // [4,300,4]
    const float* pred_masks  = (const float*)d_in[2];  // [4,300,128,128]
    const int*   tgt_ids     = (const int*)  d_in[3];  // [100]
    const float* tgt_bbox    = (const float*)d_in[4];  // [100,4]
    const float* tgt_mask    = (const float*)d_in[5];  // [100,512,512]
    float* out = (float*)d_out;                        // [4,300,100]

    _Float16* tm16  = (_Float16*)d_ws;                               // 112*16384 f16
    float*    tmsum = (float*)((char*)d_ws + (size_t)TPAD * HW * 2); // 112 f32

    hm_downsample_kernel<<<TPAD, 256, 0, stream>>>(tgt_mask, tm16, tmsum);
    hm_scalar_cost_kernel<<<NQTOT, 128, 0, stream>>>(pred_logits, pred_boxes,
                                                     tgt_ids, tgt_bbox, out);
    hm_mask_gemm_kernel<<<NQTOT / 16, 256, 0, stream>>>(pred_masks, tm16, tmsum, out);
}
